// SpikeRNN2D_71038759076512
// MI455X (gfx1250) — matrix operand order repaired and checked by reading, not compiled, not run on hardware
//
#include <hip/hip_runtime.h>

// ---------------------------------------------------------------------------
// SpikeRNN2D for MI455X (gfx1250, wave32, WMMA bf16 16x16x32)
//
// B=64, L=96, C=32, H=256, NL=2, K=3, N = B*C = 2048 independent rows.
// Each block: 16 rows x all timesteps. 16 waves/block, 1 column tile/wave,
// layer weights held as persistent WMMA B-fragments in VGPRs.
// ---------------------------------------------------------------------------

typedef __bf16 bhalf;
typedef __attribute__((ext_vector_type(16))) __bf16 v16bf;
typedef __attribute__((ext_vector_type(8)))  __bf16 v8bf;
typedef __attribute__((ext_vector_type(8)))  float  v8f;

#define BSZ   64
#define LSZ   96
#define CSZ   32
#define HSZ   256
#define NROW  (BSZ * CSZ)     // 2048
#define CH    (CSZ * HSZ)     // 8192
#define SSTR  264             // padded LDS row stride (bf16 elems) -> bank spread
#define XSTR  100             // padded input row stride (floats), halo included

union AFrag { v16bf v; v8bf h[2]; };

// fp32 -> bf16 weight conversion (runs once per launch, weights stay in L2)
__global__ void wconv_kernel(const float* __restrict__ w, bhalf* __restrict__ o, int n) {
    int i = blockIdx.x * blockDim.x + threadIdx.x;
    if (i < n) o[i] = (bhalf)w[i];
}

__device__ __forceinline__ v8f wmma_step(const bhalf* __restrict__ sb, int ln, int kq,
                                         const v16bf* __restrict__ wf, v8f acc_init) {
    v8f acc = acc_init;
#pragma unroll
    for (int ks = 0; ks < 8; ++ks) {
        // A-fragment (matches B order): lane m<16 -> K {ks*32+0..15},
        // lane m>=16 -> K {ks*32+16..31}; two contiguous 16B chunks.
        AFrag a;
        const bhalf* ap = sb + ln * SSTR + ks * 32 + kq * 16;
        a.h[0] = *(const v8bf*)(ap);
        a.h[1] = *(const v8bf*)(ap + 8);
        acc = __builtin_amdgcn_wmma_f32_16x16x32_bf16(
            /*neg_a=*/false, a.v, /*neg_b=*/false, wf[ks],
            /*c_mod=*/(short)0, acc, /*reuse_a=*/false, /*reuse_b=*/false);
    }
    return acc;
}

__global__ __launch_bounds__(512, 1)
void spikernn_kernel(const float* __restrict__ in,    // [64,96,32]
                     const float* __restrict__ wenc,  // [256,1,3]
                     const float* __restrict__ benc,  // [256]
                     const float* __restrict__ brnn,  // [2,256]
                     const bhalf* __restrict__ wbf,   // [2,256,256] bf16 (row-major [j][h])
                     float* __restrict__ out1,        // [64,96,8192]
                     float* __restrict__ out2)        // [64,8192]
{
    __shared__ __align__(64) float xs[16 * XSTR];     // input rows + zero halo
    __shared__ __align__(64) bhalf sbufA[16 * SSTR];  // conv activations (bf16)
    __shared__ __align__(64) bhalf sbufB[16 * SSTR];  // layer-0 spikes (bf16)

    const int tid  = threadIdx.x;
    const int lane = tid & 31;
    const int wave = tid >> 5;          // 0..15: one 16-wide column tile each
    const int n0   = blockIdx.x * 16;   // first of this block's 16 rows
    const int bb   = n0 >> 5;           // all 16 rows share one batch index
    const int c0   = n0 & 31;           // channel base (0 or 16)

    // ---- stage input rows into LDS (t index shifted by +1 for SAME halo) ----
    for (int idx = tid; idx < 16 * LSZ; idx += 512) {
        int m = idx & 15, t = idx >> 4;
        xs[m * XSTR + t + 1] = in[bb * (LSZ * CSZ) + t * CSZ + c0 + m];
    }
    if (tid < 16) { xs[tid * XSTR + 0] = 0.f; xs[tid * XSTR + LSZ + 1] = 0.f; }

    // ---- conv coefficients: thread owns hidden column jc = tid & 255 ----
    const int   jc = tid & 255;
    const int   mbase = (tid >> 8) * 8;              // rows mbase..mbase+7
    const float w0 = wenc[jc * 3 + 0], w1 = wenc[jc * 3 + 1], w2 = wenc[jc * 3 + 2];
    const float be = benc[jc];

    // ---- WMMA tile geometry ----
    const int ln  = lane & 15;   // A row / B,C column within tile
    const int kq  = lane >> 4;   // K-half select
    const int mhi = kq * 8;      // C-fragment row offset
    const int jb  = wave * 16;   // tile column base in H

    const float bias0 = brnn[jb + ln];
    const float bias1 = brnn[HSZ + jb + ln];

    // ---- persistent B-fragments: weights pinned in VGPRs for entire run ----
    // B 32x16 fragment, K order used here (same as A): lane n<16 -> col n, K 0..15; lane n+16 -> col n, K 16..31.
    // Column j of B is row j of W (h-contiguous): one 32B contiguous load per frag.
    v16bf wf0[8], wf1[8];
#pragma unroll
    for (int ks = 0; ks < 8; ++ks) {
        size_t off = (size_t)(jb + ln) * HSZ + ks * 32 + kq * 16;
        wf0[ks] = *(const v16bf*)(wbf + off);
        wf1[ks] = *(const v16bf*)(wbf + (size_t)HSZ * HSZ + off);
    }

    v8f vm0 = {}, vm1 = {};   // persistent LIF membranes (layer0/layer1 tile)

    __syncthreads();

    for (int t = 0; t < LSZ; ++t) {
        // ---- conv encoder: h[m][jc] = b + sum_k w_k * x[m][t-1+k] ----
#pragma unroll
        for (int r = 0; r < 8; ++r) {
            int m = mbase + r;
            float hv = be + w0 * xs[m * XSTR + t]
                          + w1 * xs[m * XSTR + t + 1]
                          + w2 * xs[m * XSTR + t + 2];
            sbufA[m * SSTR + jc] = (bhalf)hv;
        }
        __syncthreads();

        // ---- layer 0: GEMM (A = conv acts, B = W0 regs) + LIF ----
        {
            v8f zero = {};
            v8f acc = wmma_step(sbufA, ln, kq, wf0, zero);
#pragma unroll
            for (int r = 0; r < 8; ++r) {
                float cur = acc[r] + bias0;
                float vn  = vm0[r] + (cur - vm0[r]) * 0.5f;   // 1/TAU = 0.5
                float sp  = (vn >= 1.0f) ? 1.0f : 0.0f;       // V_TH = 1
                vm0[r]    = vn * (1.0f - sp);                 // hard reset
                sbufB[(r + mhi) * SSTR + jb + ln] = (bhalf)sp;
            }
        }
        __syncthreads();

        // ---- layer 1: GEMM (A = spikes, B = W1 regs) + LIF -> global ----
        {
            v8f zero = {};
            v8f acc = wmma_step(sbufB, ln, kq, wf1, zero);
            const bool last = (t == LSZ - 1);
#pragma unroll
            for (int r = 0; r < 8; ++r) {
                float cur = acc[r] + bias1;
                float vn  = vm1[r] + (cur - vm1[r]) * 0.5f;
                float sp  = (vn >= 1.0f) ? 1.0f : 0.0f;
                vm1[r]    = vn * (1.0f - sp);
                int nrow = n0 + r + mhi;
                int b    = nrow >> 5;
                int c    = nrow & 31;
                size_t o = ((size_t)(b * LSZ + t)) * CH + (size_t)c * HSZ + jb + ln;
                out1[o] = sp;
                if (last) out2[(size_t)b * CH + (size_t)c * HSZ + jb + ln] = sp;
            }
        }
        __syncthreads();
    }
}

extern "C" void kernel_launch(void* const* d_in, const int* in_sizes, int n_in,
                              void* d_out, int out_size, void* d_ws, size_t ws_size,
                              hipStream_t stream) {
    const float* in   = (const float*)d_in[0];  // inputs [64,96,32]
    const float* wenc = (const float*)d_in[1];  // w_enc  [256,1,3]
    const float* benc = (const float*)d_in[2];  // b_enc  [256]
    const float* wrnn = (const float*)d_in[3];  // w_rnn  [2,256,256]
    const float* brnn = (const float*)d_in[4];  // b_rnn  [2,256]

    bhalf* wbf  = (bhalf*)d_ws;                 // 2*256*256 bf16 = 256 KB scratch
    float* out1 = (float*)d_out;
    float* out2 = out1 + (size_t)BSZ * LSZ * CH;

    const int wn = 2 * HSZ * HSZ;               // 131072
    wconv_kernel<<<(wn + 255) / 256, 256, 0, stream>>>(wrnn, wbf, wn);

    spikernn_kernel<<<NROW / 16, 512, 0, stream>>>(in, wenc, benc, brnn, wbf, out1, out2);
}
